// SoftPrompts_46918222742276
// MI455X (gfx1250) — compile-verified, matrix operand order checked by reading, not run on hardware
//
#include <hip/hip_runtime.h>
#include <hip/hip_bf16.h>

typedef __attribute__((ext_vector_type(16))) _Float16 v16h;
typedef __attribute__((ext_vector_type(8)))  _Float16 v8h;
typedef __attribute__((ext_vector_type(4)))  _Float16 v4h;
typedef __attribute__((ext_vector_type(8)))  float    v8f;

#define BDIM 4096
#define PDIM 4096
#define KDIM 512
#define LDIM 16
#define EDIM 512
#define KSEL 4

// ---------------------------------------------------------------------------
// Kernel 1: L2-normalize rows of a (nrows x 512) fp32 matrix, scale by `sgn`
// (-1 for x so the GEMM produces -cosine directly), write f16.
// One wave32 per row, 8 waves per block.
// ---------------------------------------------------------------------------
__global__ __launch_bounds__(256)
void l2norm_to_f16(const float* __restrict__ in, _Float16* __restrict__ out,
                   float sgn)
{
    const int lane = threadIdx.x & 31;
    const int wave = threadIdx.x >> 5;
    const size_t row = (size_t)blockIdx.x * 8 + wave;

    const float4* src = (const float4*)(in + row * KDIM);
    float4 v[4];
    float sq = 0.0f;
#pragma unroll
    for (int i = 0; i < 4; ++i) {
        v[i] = src[lane + i * 32];
        sq += v[i].x * v[i].x + v[i].y * v[i].y +
              v[i].z * v[i].z + v[i].w * v[i].w;
    }
#pragma unroll
    for (int off = 16; off > 0; off >>= 1)
        sq += __shfl_xor(sq, off, 32);

    const float r = sgn * rsqrtf(fmaxf(sq, 1e-12f));

    _Float16* dst = out + row * KDIM;
#pragma unroll
    for (int i = 0; i < 4; ++i) {
        const int c = (lane + i * 32) * 4;
        v4h h;
        h[0] = (_Float16)(v[i].x * r);
        h[1] = (_Float16)(v[i].y * r);
        h[2] = (_Float16)(v[i].z * r);
        h[3] = (_Float16)(v[i].w * r);
        *(v4h*)(dst + c) = h;
    }
}

// ---------------------------------------------------------------------------
// Kernel 2: scores(B,P) = A16(B,K) * B16(P,K)^T  via v_wmma_f32_16x16x32_f16.
// A16 already holds -xn, so scores == -cosine (matches reference).
// 8 waves per block; each wave computes a 16(M) x 64(N) strip, reusing the A
// fragment across 4 N-tiles. Fragment layout per CDNA5 ISA:
//   lane m = lane&15; K-half kh = (lane>>4)*8
//   halves [0..7]  = K = kb+kh .. kb+kh+7
//   halves [8..15] = K = kb+16+kh .. kb+16+kh+7
// ---------------------------------------------------------------------------
__device__ __forceinline__ v16h load_frag(const _Float16* p)
{
    v8h lo = *(const v8h*)(p);
    v8h hi = *(const v8h*)(p + 16);
    return __builtin_shufflevector(lo, hi,
                                   0, 1, 2, 3, 4, 5, 6, 7,
                                   8, 9, 10, 11, 12, 13, 14, 15);
}

__global__ __launch_bounds__(256)
void gemm_scores(const _Float16* __restrict__ A16,   // (4096, 512)  (= -xn)
                 const _Float16* __restrict__ B16,   // (4096, 512)  (=  kn)
                 float* __restrict__ S)              // (4096, 4096)
{
    const int lane = threadIdx.x & 31;
    const int wave = threadIdx.x >> 5;
    const int mrow = (blockIdx.y * 8 + wave) * 16;   // A / S row base
    const int ncol = blockIdx.x * 64;                // S col base (= B16 row)

    const int mr = lane & 15;
    const int kh = (lane >> 4) * 8;                  // 0 or 8

    const _Float16* aptr  = A16 + (size_t)(mrow + mr)       * KDIM + kh;
    const _Float16* bptr0 = B16 + (size_t)(ncol +  0 + mr)  * KDIM + kh;
    const _Float16* bptr1 = B16 + (size_t)(ncol + 16 + mr)  * KDIM + kh;
    const _Float16* bptr2 = B16 + (size_t)(ncol + 32 + mr)  * KDIM + kh;
    const _Float16* bptr3 = B16 + (size_t)(ncol + 48 + mr)  * KDIM + kh;

    v8f acc0 = {}, acc1 = {}, acc2 = {}, acc3 = {};

#pragma unroll 2
    for (int kb = 0; kb < KDIM; kb += 32) {
        v16h a  = load_frag(aptr  + kb);
        v16h b0 = load_frag(bptr0 + kb);
        v16h b1 = load_frag(bptr1 + kb);
        v16h b2 = load_frag(bptr2 + kb);
        v16h b3 = load_frag(bptr3 + kb);

        acc0 = __builtin_amdgcn_wmma_f32_16x16x32_f16(false, a, false, b0,
                                                      (short)0, acc0, false, false);
        acc1 = __builtin_amdgcn_wmma_f32_16x16x32_f16(false, a, false, b1,
                                                      (short)0, acc1, false, false);
        acc2 = __builtin_amdgcn_wmma_f32_16x16x32_f16(false, a, false, b2,
                                                      (short)0, acc2, false, false);
        acc3 = __builtin_amdgcn_wmma_f32_16x16x32_f16(false, a, false, b3,
                                                      (short)0, acc3, false, false);
    }

    // C/D layout: lanes 0-15 hold M = r (VGPR r), lanes 16-31 hold M = 8 + r,
    // N = lane & 15.
    const int mbase = mrow + (lane >> 4) * 8;
#pragma unroll
    for (int r = 0; r < 8; ++r) {
        float* p = S + (size_t)(mbase + r) * PDIM + ncol + mr;
        p[0]  = acc0[r];
        p[16] = acc1[r];
        p[32] = acc2[r];
        p[48] = acc3[r];
    }
}

// ---------------------------------------------------------------------------
// Kernel 3: top-4 (values descending, ties -> lower index) over each row of
// scores. One 256-thread block per row; 4 rounds of block argmax.
// Thread t owns columns t + i*256, i in [0,16).
// ---------------------------------------------------------------------------
__global__ __launch_bounds__(256)
void topk4(const float* __restrict__ S, int* __restrict__ idx)
{
    __shared__ float sv[256];
    __shared__ int   si[256];
    const int t = threadIdx.x;
    const float* s = S + (size_t)blockIdx.x * PDIM;

    float vals[16];
#pragma unroll
    for (int i = 0; i < 16; ++i) vals[i] = s[t + i * 256];

    for (int round = 0; round < KSEL; ++round) {
        float bv = vals[0];
        int   bs = 0;
#pragma unroll
        for (int i = 1; i < 16; ++i)
            if (vals[i] > bv) { bv = vals[i]; bs = i; }   // '>' keeps lower col
        sv[t] = bv;
        si[t] = t + bs * 256;
        __syncthreads();
        for (int off = 128; off > 0; off >>= 1) {
            if (t < off) {
                const float v2 = sv[t + off];
                const int   i2 = si[t + off];
                if (v2 > sv[t] || (v2 == sv[t] && i2 < si[t])) {
                    sv[t] = v2; si[t] = i2;
                }
            }
            __syncthreads();
        }
        const int wi = si[0];
        if (t == 0) idx[blockIdx.x * KSEL + round] = wi;
        if ((wi & 255) == t) vals[wi >> 8] = -__builtin_inff();
        __syncthreads();
    }
}

// ---------------------------------------------------------------------------
// Kernel 4: out[b, j*16 + l, :] = prompt_values[idx[b*4+j], l, :]
// Flat float4 copy; 128 float4 per 2KB row. HBM-bound (~1 GB traffic).
// ---------------------------------------------------------------------------
__global__ __launch_bounds__(256)
void gather_out(const float4* __restrict__ pv, const int* __restrict__ idx,
                float4* __restrict__ out)
{
    const size_t g      = (size_t)blockIdx.x * 256 + threadIdx.x;
    const int    within = (int)(g & 127);        // float4 within E=512 row
    const size_t rowid  = g >> 7;                // b*64 + j*16 + l
    const int    l16    = (int)(rowid & 15);
    const size_t bj     = rowid >> 4;            // b*4 + j
    const int    p      = idx[bj];
    out[g] = pv[(((size_t)p * LDIM + l16) << 7) + within];
}

// ---------------------------------------------------------------------------
extern "C" void kernel_launch(void* const* d_in, const int* in_sizes, int n_in,
                              void* d_out, int out_size, void* d_ws, size_t ws_size,
                              hipStream_t stream)
{
    const float* x  = (const float*)d_in[0];   // (4096, 64, 8)
    const float* pk = (const float*)d_in[1];   // (4096, 512)
    const float* pv = (const float*)d_in[2];   // (4096, 16, 512)
    float* out = (float*)d_out;                // (4096, 64, 512)

    char* ws = (char*)d_ws;
    _Float16* xn16 = (_Float16*)(ws);                          //  4 MB (= -xn)
    _Float16* kn16 = (_Float16*)(ws + ((size_t)4  << 20));     //  4 MB
    float*    S    = (float*)   (ws + ((size_t)8  << 20));     // 64 MB
    int*      idx  = (int*)     (ws + ((size_t)72 << 20));     // 64 KB

    (void)in_sizes; (void)n_in; (void)out_size; (void)ws_size;

    l2norm_to_f16<<<BDIM / 8, 256, 0, stream>>>(x, xn16, -1.0f);
    l2norm_to_f16<<<PDIM / 8, 256, 0, stream>>>(pk, kn16, 1.0f);

    gemm_scores<<<dim3(PDIM / 64, BDIM / 128), 256, 0, stream>>>(xn16, kn16, S);

    topk4<<<BDIM, 256, 0, stream>>>(S, idx);

    const size_t total_f4 = (size_t)BDIM * KSEL * LDIM * EDIM / 4;  // 33,554,432
    gather_out<<<(unsigned)(total_f4 / 256), 256, 0, stream>>>(
        (const float4*)pv, idx, (float4*)out);
}
